// RoPETimeSeriesAttention_24395414241948
// MI455X (gfx1250) — matrix operand order repaired and checked by reading, not compiled, not run on hardware
//
#include <hip/hip_runtime.h>

#define D_MODEL 1024
#define N_HEADS 16
#define HEAD_DIM 64
#define BATCH 2
#define SEQ 2048
#define BSROWS (BATCH * SEQ)   // 4096

typedef __attribute__((ext_vector_type(16))) _Float16 v16h;
typedef __attribute__((ext_vector_type(8)))  _Float16 v8h;
typedef __attribute__((ext_vector_type(8)))  float    v8f;

__device__ __forceinline__ int lane_id() { return (int)(threadIdx.x & 31u); }

__device__ __forceinline__ v8f zero_v8f() {
    v8f z;
#pragma unroll
    for (int i = 0; i < 8; ++i) z[i] = 0.0f;
    return z;
}

// ---------------------------------------------------------------------------
// CDNA5 async copy: GLOBAL_LOAD_ASYNC_TO_LDS_B128 (GVS addressing, ASYNCcnt).
// lds_ptr: generic pointer into __shared__; low 32 bits are the LDS byte
// address (LDS aperture maps addr[31:0] -> LDS offset per ISA 10.2).
// ---------------------------------------------------------------------------
__device__ __forceinline__ void async_copy_b128(void* lds_ptr, const void* gbase,
                                                unsigned byte_off) {
    unsigned ldso = (unsigned)(uintptr_t)lds_ptr;
    asm volatile("global_load_async_to_lds_b128 %0, %1, %2"
                 :
                 : "v"(ldso), "v"(byte_off), "s"(gbase)
                 : "memory");
}

__device__ __forceinline__ void wait_async0() {
#if __has_builtin(__builtin_amdgcn_s_wait_asynccnt)
    __builtin_amdgcn_s_wait_asynccnt(0);
#else
    asm volatile("s_wait_asynccnt 0x0" ::: "memory");
#endif
}

// A fragment 16x32 f16 (M x K) from a row-major LDS tile (ldh halves/row).
// ISA layout: lanes 0-15 -> M=lane, K {0..7} in v0-3, {16..23} in v4-7;
//             lanes 16-31 -> M=lane-16, K {8..15} and {24..31}.
__device__ __forceinline__ v16h load_a_frag(const _Float16* tile, int ldh) {
    int l = lane_id();
    const _Float16* p = tile + (l & 15) * ldh + ((l >> 4) << 3);
    v8h lo = *(const v8h*)(p);        // K = kg .. kg+7
    v8h hi = *(const v8h*)(p + 16);   // K = kg+16 .. kg+23
    v16h a;
#pragma unroll
    for (int i = 0; i < 8; ++i) { a[i] = lo[i]; a[8 + i] = hi[i]; }
    return a;
}

// B fragment 32x16 f16 (K x N), sourced from a Bt tile stored [N][K].
__device__ __forceinline__ v16h load_b_frag(const _Float16* bt, int ldh) {
    int l = lane_id();
    const _Float16* p = bt + (l & 15) * ldh + ((l >> 4) << 3);
    v8h lo = *(const v8h*)(p);
    v8h hi = *(const v8h*)(p + 16);
    v16h b;
#pragma unroll
    for (int i = 0; i < 8; ++i) { b[i] = lo[i]; b[8 + i] = hi[i]; }
    return b;
}

__device__ __forceinline__ v8f wmma_f16(v16h a, v16h b, v8f c) {
    return __builtin_amdgcn_wmma_f32_16x16x32_f16(false, a, false, b, (short)0, c,
                                                  false, false);
}

// ---------------------------------------------------------------------------
// Kernel 1: fused Q/K/V projection (+bias), RoPE on Q/K, V stored transposed.
// grid = (BSROWS/128, D_MODEL/64, 3), block = 256 (8 waves).
// ---------------------------------------------------------------------------
__global__ __launch_bounds__(256) void qkv_proj_rope_kernel(
    const float* __restrict__ x,
    const float* __restrict__ Wq, const float* __restrict__ bq,
    const float* __restrict__ Wk, const float* __restrict__ bk,
    const float* __restrict__ Wv, const float* __restrict__ bv,
    _Float16* __restrict__ qws, _Float16* __restrict__ kws,
    _Float16* __restrict__ vtws)
{
    const int mz = blockIdx.z;  // 0=Q, 1=K, 2=V
    const float* W   = (mz == 0) ? Wq : (mz == 1) ? Wk : Wv;
    const float* bia = (mz == 0) ? bq : (mz == 1) ? bk : bv;

    const int m0 = blockIdx.x * 128;
    const int n0 = blockIdx.y * 64;

    __shared__ __align__(128) _Float16 At[128 * 48];  // x tile, stride 48
    __shared__ __align__(128) _Float16 Bt[64 * 48];   // W tile (natural [e][d])

    const int tid  = (int)threadIdx.x;
    const int wave = tid >> 5;
    const int l    = tid & 31;

    v8f acc[4];
#pragma unroll
    for (int t = 0; t < 4; ++t) acc[t] = zero_v8f();

    for (int k0 = 0; k0 < D_MODEL; k0 += 32) {
        __syncthreads();
#pragma unroll
        for (int i = 0; i < 16; ++i) {            // 128x32 f32 -> f16
            int idx = i * 256 + tid;
            int r = idx >> 5, c = idx & 31;
            At[r * 48 + c] = (_Float16)x[(size_t)(m0 + r) * D_MODEL + k0 + c];
        }
#pragma unroll
        for (int i = 0; i < 8; ++i) {             // 64x32 f32 -> f16
            int idx = i * 256 + tid;
            int r = idx >> 5, c = idx & 31;
            Bt[r * 48 + c] = (_Float16)W[(size_t)(n0 + r) * D_MODEL + k0 + c];
        }
        __syncthreads();

        v16h a = load_a_frag(At + wave * 16 * 48, 48);
#pragma unroll
        for (int t = 0; t < 4; ++t) {
            v16h b = load_b_frag(Bt + t * 16 * 48, 48);
            acc[t] = wmma_f16(a, b, acc[t]);
        }
    }

    const int col    = l & 15;
    const int rbase  = (l >> 4) * 8;
    const int gmbase = m0 + wave * 16;
    const int h      = n0 >> 6;

    if (mz < 2) {
        _Float16* outp = (mz == 0) ? qws : kws;
#pragma unroll
        for (int t = 0; t < 2; ++t) {
            int   hd   = t * 16 + col;                       // [0, 32)
            float freq = __powf(10000.0f, -(float)hd * (1.0f / 32.0f));
            float b0   = bia[n0 + t * 16 + col];
            float b1   = bia[n0 + 32 + t * 16 + col];
#pragma unroll
            for (int r = 0; r < 8; ++r) {
                int gm = gmbase + rbase + r;
                int bb = gm >> 11;          // / SEQ
                int s  = gm & (SEQ - 1);
                float ang = (float)s * freq;
                float sn, cs;
                __sincosf(ang, &sn, &cs);
                float v0 = acc[t][r] + b0;
                float v1 = acc[t + 2][r] + b1;
                float y0 = v0 * cs - v1 * sn;
                float y1 = v1 * cs + v0 * sn;
                size_t base = ((size_t)(bb * N_HEADS + h) * SEQ + s) * HEAD_DIM;
                outp[base + hd]      = (_Float16)y0;
                outp[base + hd + 32] = (_Float16)y1;
            }
        }
    } else {
#pragma unroll
        for (int t = 0; t < 4; ++t) {
            int   hd = t * 16 + col;
            float bb_ = bia[n0 + t * 16 + col];
#pragma unroll
            for (int r = 0; r < 8; ++r) {
                int gm = gmbase + rbase + r;
                int bb = gm >> 11;
                int s  = gm & (SEQ - 1);
                float v = acc[t][r] + bb_;
                vtws[((size_t)(bb * N_HEADS + h) * HEAD_DIM + hd) * SEQ + s] =
                    (_Float16)v;
            }
        }
    }
}

// ---------------------------------------------------------------------------
// Kernel 2: flash attention per (b,h), async double-buffered K/V staging.
// grid = (SEQ/128, BATCH*N_HEADS).
// ---------------------------------------------------------------------------
__global__ __launch_bounds__(256) void flash_attn_kernel(
    const _Float16* __restrict__ qws, const _Float16* __restrict__ kws,
    const _Float16* __restrict__ vtws, _Float16* __restrict__ aows)
{
    const int head = blockIdx.y;            // b*N_HEADS + h
    const int m0   = blockIdx.x * 128;
    const int tid  = (int)threadIdx.x;
    const int wave = tid >> 5;
    const int l    = tid & 31;

    __shared__ __align__(128) _Float16 Qs[128 * 72];       // stride 72 (144B)
    __shared__ __align__(128) _Float16 Ks[2][32 * 72];     // double-buffered keys
    __shared__ __align__(128) _Float16 Vs[2][64 * 48];     // double-buffered Vt
    __shared__ __align__(128) _Float16 Ps[8][16 * 48];     // per-wave P buffer

    const _Float16* qg = qws  + (size_t)head * SEQ * HEAD_DIM;
    const _Float16* kg = kws  + (size_t)head * SEQ * HEAD_DIM;
    const _Float16* vg = vtws + (size_t)head * HEAD_DIM * SEQ;

    {   // stage d 0..15 and 32..47 of 128 Q rows via async DMA to LDS (2 x B128 per thread)
        int r = tid >> 1, part = (tid & 1) * 32;
        unsigned goff = (unsigned)(((m0 + r) * HEAD_DIM + part) * 2);
        async_copy_b128(Qs + r * 72 + part,     qg, goff);
        async_copy_b128(Qs + r * 72 + part + 8, qg, goff + 16);
    }

    // issue first K/V chunk while Q DMA is in flight
    {
        int r = tid >> 3, seg = (tid & 7) * 8;
        async_copy_b128(&Ks[0][r * 72 + seg], kg,
                        (unsigned)((r * HEAD_DIM + seg) * 2));
        int rv = tid >> 2, sv = (tid & 3) * 8;
        async_copy_b128(&Vs[0][rv * 48 + sv], vg,
                        (unsigned)((rv * SEQ + sv) * 2));
    }
    wait_async0();
    __syncthreads();

    v16h qa0 = load_a_frag(Qs + wave * 16 * 72, 72);       // d 0..31
    v16h qa1 = load_a_frag(Qs + wave * 16 * 72 + 32, 72);  // d 32..63

    float mst[8], lst[8];
    v8f o[4];
#pragma unroll
    for (int r = 0; r < 8; ++r) { mst[r] = -1e30f; lst[r] = 0.0f; }
#pragma unroll
    for (int t = 0; t < 4; ++t) o[t] = zero_v8f();

    const float scale = 0.125f;  // 1/sqrt(64)

    for (int j = 0; j < SEQ; j += 32) {
        const int cur = (j >> 5) & 1;
        if (j) {                 // chunk j's DMA was issued last iteration
            wait_async0();
            __syncthreads();
        }
        if (j + 32 < SEQ) {      // issue next chunk into the other buffer
            int jn = j + 32;
            int r = tid >> 3, seg = (tid & 7) * 8;
            async_copy_b128(&Ks[cur ^ 1][r * 72 + seg], kg,
                            (unsigned)(((jn + r) * HEAD_DIM + seg) * 2));
            int rv = tid >> 2, sv = (tid & 3) * 8;
            async_copy_b128(&Vs[cur ^ 1][rv * 48 + sv], vg,
                            (unsigned)((rv * SEQ + jn + sv) * 2));
            if (j + 64 < SEQ)    // keep L2 warm further ahead
                __builtin_prefetch(kg + (size_t)(j + 64) * HEAD_DIM, 0, 0);
        }

        const _Float16* Kb = &Ks[cur][0];
        const _Float16* Vb = &Vs[cur][0];

        // scores: 16 q rows x 32 keys (two 16x16 fragments)
        v8f s0 = zero_v8f(), s1 = zero_v8f();
        v16h kb0  = load_b_frag(Kb, 72);                 // keys j..j+15,  d 0..31
        v16h kb0b = load_b_frag(Kb + 32, 72);            // keys j..j+15,  d 32..63
        v16h kb1  = load_b_frag(Kb + 16 * 72, 72);       // keys j+16..31, d 0..31
        v16h kb1b = load_b_frag(Kb + 16 * 72 + 32, 72);  // keys j+16..31, d 32..63
        s0 = wmma_f16(qa0, kb0, s0);
        s0 = wmma_f16(qa1, kb0b, s0);
        s1 = wmma_f16(qa0, kb1, s1);
        s1 = wmma_f16(qa1, kb1b, s1);

        // online softmax (row groups: 16 lanes share one set of 8 rows)
#pragma unroll
        for (int r = 0; r < 8; ++r) {
            float v0 = s0[r] * scale, v1 = s1[r] * scale;
            float mx = fmaxf(v0, v1);
#pragma unroll
            for (int d = 1; d < 16; d <<= 1) mx = fmaxf(mx, __shfl_xor(mx, d, 32));
            float mnew = fmaxf(mst[r], mx);
            float e0 = __expf(v0 - mnew);
            float e1 = __expf(v1 - mnew);
            s0[r] = e0; s1[r] = e1;
            float sum = e0 + e1;
#pragma unroll
            for (int d = 1; d < 16; d <<= 1) sum += __shfl_xor(sum, d, 32);
            float corr = __expf(mst[r] - mnew);
            lst[r] = lst[r] * corr + sum;
            mst[r] = mnew;
            o[0][r] *= corr; o[1][r] *= corr; o[2][r] *= corr; o[3][r] *= corr;
        }

        // re-fragment P (C-layout -> A-layout) through per-wave LDS
        _Float16* P = &Ps[wave][0];
        {
            int col = l & 15, rhi = (l >> 4) * 8;
#pragma unroll
            for (int r = 0; r < 8; ++r) {
                P[(rhi + r) * 48 + col]      = (_Float16)s0[r];
                P[(rhi + r) * 48 + col + 16] = (_Float16)s1[r];
            }
        }
        v16h pa = load_a_frag(P, 48);
#pragma unroll
        for (int t = 0; t < 4; ++t) {
            v16h vb = load_b_frag(Vb + t * 16 * 48, 48);
            o[t] = wmma_f16(pa, vb, o[t]);
        }
        __syncthreads();   // all reads of buf[cur] done before its next refill
    }

    // epilogue: normalize, store f16 to [B, S, H*Hd]
    const int col = l & 15, rhi = (l >> 4) * 8;
    const int b = head >> 4, h = head & 15;
#pragma unroll
    for (int r = 0; r < 8; ++r) {
        int s = m0 + wave * 16 + rhi + r;
        float inv = 1.0f / lst[r];
        size_t base = ((size_t)(b * SEQ + s)) * D_MODEL + h * HEAD_DIM;
#pragma unroll
        for (int t = 0; t < 4; ++t)
            aows[base + t * 16 + col] = (_Float16)(o[t][r] * inv);
    }
}

// ---------------------------------------------------------------------------
// Kernel 3: output projection (async A-tile staging).
// grid = (BSROWS/128, D_MODEL/64).
// ---------------------------------------------------------------------------
__global__ __launch_bounds__(256) void out_proj_kernel(
    const _Float16* __restrict__ aows, const float* __restrict__ Wo,
    const float* __restrict__ bo, float* __restrict__ out)
{
    const int m0 = blockIdx.x * 128;
    const int n0 = blockIdx.y * 64;

    __shared__ __align__(128) _Float16 At[128 * 48];
    __shared__ __align__(128) _Float16 Bt[64 * 48];

    const int tid = (int)threadIdx.x;
    const int wave = tid >> 5;
    const int l = tid & 31;

    v8f acc[4];
#pragma unroll
    for (int t = 0; t < 4; ++t) acc[t] = zero_v8f();

    for (int k0 = 0; k0 < D_MODEL; k0 += 32) {
        __syncthreads();
        {   // A tile: f16 source -> async DMA to LDS (2 x B128 per thread)
            int r = tid >> 1, seg = (tid & 1) * 16;
            unsigned goff = (unsigned)(((m0 + r) * D_MODEL + k0 + seg) * 2);
            async_copy_b128(At + r * 48 + seg,     aows, goff);
            async_copy_b128(At + r * 48 + seg + 8, aows, goff + 16);
        }
#pragma unroll
        for (int i = 0; i < 8; ++i) {   // Wo tile 64x32 f32 -> f16
            int idx = i * 256 + tid;
            int r = idx >> 5, c = idx & 31;
            Bt[r * 48 + c] = (_Float16)Wo[(size_t)(n0 + r) * D_MODEL + k0 + c];
        }
        wait_async0();
        __syncthreads();

        v16h a = load_a_frag(At + wave * 16 * 48, 48);
#pragma unroll
        for (int t = 0; t < 4; ++t) {
            v16h b = load_b_frag(Bt + t * 16 * 48, 48);
            acc[t] = wmma_f16(a, b, acc[t]);
        }
    }

    const int col = l & 15, rhi = (l >> 4) * 8;
#pragma unroll
    for (int t = 0; t < 4; ++t) {
        float bb = bo[n0 + t * 16 + col];
#pragma unroll
        for (int r = 0; r < 8; ++r) {
            int gm = m0 + wave * 16 + rhi + r;
            out[(size_t)gm * D_MODEL + n0 + t * 16 + col] = acc[t][r] + bb;
        }
    }
}

// ---------------------------------------------------------------------------
extern "C" void kernel_launch(void* const* d_in, const int* in_sizes, int n_in,
                              void* d_out, int out_size, void* d_ws, size_t ws_size,
                              hipStream_t stream) {
    const float* x  = (const float*)d_in[0];
    const float* Wq = (const float*)d_in[1];
    const float* bq = (const float*)d_in[2];
    const float* Wk = (const float*)d_in[3];
    const float* bk = (const float*)d_in[4];
    const float* Wv = (const float*)d_in[5];
    const float* bv = (const float*)d_in[6];
    const float* Wo = (const float*)d_in[7];
    const float* bo = (const float*)d_in[8];
    float* out = (float*)d_out;

    const size_t segBytes = (size_t)BSROWS * D_MODEL * sizeof(_Float16);  // 8 MB
    char* wsb = (char*)d_ws;
    _Float16* qws  = (_Float16*)(wsb + 0 * segBytes);   // [B,H,S,Hd]
    _Float16* kws  = (_Float16*)(wsb + 1 * segBytes);   // [B,H,S,Hd]
    _Float16* vtws = (_Float16*)(wsb + 2 * segBytes);   // [B,H,Hd,S]
    _Float16* aows = (_Float16*)(wsb + 3 * segBytes);   // [B,S,D]
    (void)ws_size; (void)in_sizes; (void)n_in; (void)out_size;

    dim3 g1(BSROWS / 128, D_MODEL / 64, 3);
    qkv_proj_rope_kernel<<<g1, 256, 0, stream>>>(x, Wq, bq, Wk, bk, Wv, bv,
                                                 qws, kws, vtws);

    dim3 g2(SEQ / 128, BATCH * N_HEADS);
    flash_attn_kernel<<<g2, 256, 0, stream>>>(qws, kws, vtws, aows);

    dim3 g3(BSROWS / 128, D_MODEL / 64);
    out_proj_kernel<<<g3, 256, 0, stream>>>(aows, Wo, bo, out);
}
